// EnhancedQSelfAttention_44762149159707
// MI455X (gfx1250) — matrix operand order repaired and checked by reading, not compiled, not run on hardware
//
#include <hip/hip_runtime.h>
#include <hip/hip_bf16.h>
#include <stdint.h>

// Problem sizes (fixed by the reference)
#define B_DIM 8
#define C_DIM 512
#define N_TOK 4096
#define CK_DIM 64
#define MB 128  // m-rows staged per flash round (512x128 bf16 V slab = 128KB LDS)

typedef __attribute__((ext_vector_type(16))) __bf16 v16bf;
typedef __attribute__((ext_vector_type(8)))  __bf16 v8bf;
typedef __attribute__((ext_vector_type(8)))  float  v8f;
typedef __attribute__((ext_vector_type(8)))  float  v8fv;
typedef __attribute__((ext_vector_type(4)))  unsigned int u32x4;
typedef __attribute__((ext_vector_type(8)))  int i32x8;
typedef __attribute__((ext_vector_type(4)))  int i32x4;

#define GAS __attribute__((address_space(1)))

// Explicit global-address-space accessors: force global_load_* / global_store_*
// (FLAT ops would tie up DScnt and the LDS pipe alongside LOADcnt).
static __device__ __forceinline__ v8bf  gload_v8bf (const __bf16* p) { return *(const GAS v8bf*) (uintptr_t)p; }
static __device__ __forceinline__ v16bf gload_v16bf(const __bf16* p) { return *(const GAS v16bf*)(uintptr_t)p; }
static __device__ __forceinline__ v8fv  gload_v8f  (const float*  p) { return *(const GAS v8fv*) (uintptr_t)p; }
static __device__ __forceinline__ float gload_f32  (const float*  p) { return *(const GAS float*)(uintptr_t)p; }
static __device__ __forceinline__ void  gstore_bf16(__bf16* p, __bf16 v) { *(GAS __bf16*)(uintptr_t)p = v; }
static __device__ __forceinline__ void  gstore_f32 (float*  p, float  v) { *(GAS float*) (uintptr_t)p = v; }

static __device__ __forceinline__ v8f wmma_bf16(v16bf a, v16bf b, v8f c) {
  // D = A(16x32 bf16) * B(32x16 bf16) + C(16x16 f32)
  return __builtin_amdgcn_wmma_f32_16x16x32_bf16(false, a, false, b, (short)0, c, false, false);
}

// A-operand (16x32, 16-bit): per lane, one contiguous 16-element run:
//   e=0..15 -> k = k0 + 16*half + e
static __device__ __forceinline__ v16bf load_a_g(const __bf16* base /* +k0+16*half, global */) {
  v8bf lo = gload_v8bf(base);
  v8bf hi = gload_v8bf(base + 8);
  return __builtin_shufflevector(lo, hi, 0, 1, 2, 3, 4, 5, 6, 7, 8, 9, 10, 11, 12, 13, 14, 15);
}
static __device__ __forceinline__ v16bf load_a_l(const __bf16* base /* +k0+16*half, LDS */) {
  v8bf lo = *(const v8bf*)(base);
  v8bf hi = *(const v8bf*)(base + 8);
  return __builtin_shufflevector(lo, hi, 0, 1, 2, 3, 4, 5, 6, 7, 8, 9, 10, 11, 12, 13, 14, 15);
}
static __device__ __forceinline__ v16bf load_a_gf32(const float* base /* +k0+16*half, global */) {
  v8fv lo = gload_v8f(base);
  v8fv hi = gload_v8f(base + 8);
  v16bf a;
#pragma unroll
  for (int e = 0; e < 8; ++e) { a[e] = (__bf16)lo[e]; a[8 + e] = (__bf16)hi[e]; }
  return a;
}

// ---------------------------------------------------------------------------
// K1: 1x1 conv projections.  One block per (n-tile, batch); the 16x512 x-tile
// is staged once in LDS (bf16), then 8 waves split the 40 output tiles
// (Q:4, K:4, V:32), amortizing each A build over 5 WMMAs.
// Q is pre-scaled by ck^-0.5.  V is written TRANSPOSED: Vbt[B][C][N].
// ---------------------------------------------------------------------------
__global__ void proj_kernel(const float* __restrict__ x,
                            const float* __restrict__ wq, const float* __restrict__ bq,
                            const float* __restrict__ wk, const float* __restrict__ bk,
                            const float* __restrict__ wv, const float* __restrict__ bv,
                            __bf16* __restrict__ Qb, __bf16* __restrict__ Kb,
                            __bf16* __restrict__ Vbt) {
  __shared__ __bf16 xt[16][C_DIM];  // [token][channel], 16KB

  const int tid  = threadIdx.x;
  const int w    = tid >> 5;
  const int lane = tid & 31;
  const int half = lane >> 4, l16 = lane & 15;
  const int n0 = blockIdx.x * 16;
  const int b  = blockIdx.y;

  // Stage + transpose x[b][:, n0:n0+16] -> xt[n][c] (bf16)
  const float* xb = x + (size_t)b * C_DIM * N_TOK;
  for (int idx = tid; idx < 16 * C_DIM; idx += 256) {
    int c = idx >> 4, nl = idx & 15;
    xt[nl][c] = (__bf16)gload_f32(xb + (size_t)c * N_TOK + n0 + nl);
  }
  __syncthreads();

  // This wave's 5 output tiles (z = w*5 .. w*5+4; z<4 Q, z<8 K, else V)
  const float* Wj[5]; const float* Bj[5]; __bf16* Oj[5];
  int o0j[5]; float scj[5]; bool isVj[5];
#pragma unroll
  for (int j = 0; j < 5; ++j) {
    int z = w * 5 + j;
    if (z < 4)      { Wj[j] = wq; Bj[j] = bq; Oj[j] = Qb;  o0j[j] = z * 16;       scj[j] = 0.125f; isVj[j] = false; }
    else if (z < 8) { Wj[j] = wk; Bj[j] = bk; Oj[j] = Kb;  o0j[j] = (z - 4) * 16; scj[j] = 1.0f;   isVj[j] = false; }
    else            { Wj[j] = wv; Bj[j] = bv; Oj[j] = Vbt; o0j[j] = (z - 8) * 16; scj[j] = 1.0f;   isVj[j] = true;  }
  }

  v8f acc[5] = {};
  for (int c0 = 0; c0 < C_DIM; c0 += 32) {
    v16bf a = load_a_l(&xt[l16][c0 + 16 * half]);      // A: tokens x channels
#pragma unroll
    for (int j = 0; j < 5; ++j) {
      // B: W^T chunk, contiguous 16 channels per lane starting at c0+16*half
      const float* wr = Wj[j] + (size_t)(o0j[j] + l16) * C_DIM + c0 + 16 * half;
      v8fv lo = gload_v8f(wr);
      v8fv hi = gload_v8f(wr + 8);
      v16bf bm;
#pragma unroll
      for (int e = 0; e < 8; ++e) { bm[e] = (__bf16)lo[e]; bm[8 + e] = (__bf16)hi[e]; }
      acc[j] = wmma_bf16(a, bm, acc[j]);
    }
  }

#pragma unroll
  for (int j = 0; j < 5; ++j) {
    const int o = o0j[j] + l16;          // D column = out channel
    const float bias_o = gload_f32(Bj[j] + o);
    if (isVj[j]) {
      __bf16* ob = Oj[j] + (size_t)b * C_DIM * N_TOK;   // [C][N]
#pragma unroll
      for (int r = 0; r < 8; ++r) {
        int row = n0 + r + 8 * half;     // D row = token
        gstore_bf16(ob + (size_t)o * N_TOK + row, (__bf16)((acc[j][r] + bias_o) * scj[j]));
      }
    } else {
      __bf16* ob = Oj[j] + (size_t)b * N_TOK * CK_DIM;  // [N][CK]
#pragma unroll
      for (int r = 0; r < 8; ++r) {
        int row = n0 + r + 8 * half;
        gstore_bf16(ob + (size_t)row * CK_DIM + o, (__bf16)((acc[j][r] + bias_o) * scj[j]));
      }
    }
  }
}

// ---------------------------------------------------------------------------
// K2: per-row max/min of S = Q*K^T (scale folded into Q).  One wave per
// 16-row n-tile, sweeping all 256 m-tiles with WMMA.
// ---------------------------------------------------------------------------
__global__ void rowstat_kernel(const __bf16* __restrict__ Qb, const __bf16* __restrict__ Kb,
                               float* __restrict__ rowmax, float* __restrict__ rowmin) {
  const int lane = threadIdx.x & 31;
  const int half = lane >> 4, l16 = lane & 15;
  const int n0 = blockIdx.x * 16;
  const int b  = blockIdx.y;
  const __bf16* Q = Qb + (size_t)b * N_TOK * CK_DIM;
  const __bf16* K = Kb + (size_t)b * N_TOK * CK_DIM;

  const __bf16* qrow = Q + (size_t)(n0 + l16) * CK_DIM;
  v16bf a0 = load_a_g(qrow + 16 * half);
  v16bf a1 = load_a_g(qrow + 32 + 16 * half);

  float vmax[8], vmin[8];
#pragma unroll
  for (int r = 0; r < 8; ++r) { vmax[r] = -3.0e38f; vmin[r] = 3.0e38f; }

  for (int m0 = 0; m0 < N_TOK; m0 += 16) {
    const __bf16* krow = K + (size_t)(m0 + l16) * CK_DIM;
    v16bf b0 = gload_v16bf(krow + 16 * half);            // contiguous in ck
    v16bf b1 = gload_v16bf(krow + 32 + 16 * half);
    v8f s = {};
    s = wmma_bf16(a0, b0, s);
    s = wmma_bf16(a1, b1, s);
#pragma unroll
    for (int r = 0; r < 8; ++r) {
      vmax[r] = fmaxf(vmax[r], s[r]);
      vmin[r] = fminf(vmin[r], s[r]);
    }
  }
#pragma unroll
  for (int m = 1; m < 16; m <<= 1) {
#pragma unroll
    for (int r = 0; r < 8; ++r) {
      vmax[r] = fmaxf(vmax[r], __shfl_xor(vmax[r], m, 32));
      vmin[r] = fminf(vmin[r], __shfl_xor(vmin[r], m, 32));
    }
  }
  if (l16 == 0) {
#pragma unroll
    for (int r = 0; r < 8; ++r) {
      int row = n0 + r + 8 * half;
      gstore_f32(rowmax + (size_t)b * N_TOK + row, vmax[r]);
      gstore_f32(rowmin + (size_t)b * N_TOK + row, vmin[r]);
    }
  }
}

// ---------------------------------------------------------------------------
// K3: global quantization params. emax==1 exactly; emin = exp(min(rowmin-rowmax)).
// ---------------------------------------------------------------------------
__global__ void quantparam_kernel(const float* __restrict__ rowmax,
                                  const float* __restrict__ rowmin,
                                  float* __restrict__ params) {
  __shared__ float red[256];
  float g = 3.0e38f;
  for (int i = threadIdx.x; i < B_DIM * N_TOK; i += 256)
    g = fminf(g, gload_f32(rowmin + i) - gload_f32(rowmax + i));
  red[threadIdx.x] = g;
  __syncthreads();
  for (int s = 128; s > 0; s >>= 1) {
    if ((int)threadIdx.x < s) red[threadIdx.x] = fminf(red[threadIdx.x], red[threadIdx.x + s]);
    __syncthreads();
  }
  if (threadIdx.x == 0) {
    float emin  = __builtin_exp2f(red[0] * 1.44269504f);
    float scale = 255.0f / (1.0f - emin);
    gstore_f32(params + 0, scale);
    gstore_f32(params + 1, scale * emin);   // zero_point
    gstore_f32(params + 2, 1.0f / scale);
  }
}

// ---------------------------------------------------------------------------
// K4: flash attention with quantized softmax.  256 threads (8 waves) per
// 16-row n-tile.  Per round: TDM stages V^T[0:512][m0:m0+128] into LDS
// ([c][m], so B operands are contiguous) while the 8 waves compute the
// quantized-P 16x128 tile; then each wave WMMAs its 64-column strip.
// ---------------------------------------------------------------------------
#define SMEM_V (C_DIM * MB * 2)
#define SMEM_P (16 * MB * 2)

__global__ void flash_kernel(const __bf16* __restrict__ Qb, const __bf16* __restrict__ Kb,
                             const __bf16* __restrict__ Vbt, const float* __restrict__ rowmax,
                             const float* __restrict__ params, __bf16* __restrict__ attO) {
  extern __shared__ char smem[];
  __bf16* v_lds  = (__bf16*)smem;                      // [C_DIM][MB]
  __bf16* p_lds  = (__bf16*)(smem + SMEM_V);           // [16][MB]
  float*  rs_lds = (float*)(smem + SMEM_V + SMEM_P);   // [8 waves][16 rows]
  float*  inv_rs = rs_lds + 8 * 16;

  const int tid  = threadIdx.x;
  const int w    = tid >> 5;
  const int lane = tid & 31;
  const int half = lane >> 4, l16 = lane & 15;
  const int n0 = blockIdx.x * 16;
  const int b  = blockIdx.y;

  const float scale = gload_f32(params + 0);
  const float zp = gload_f32(params + 1);
  const float inv_scale = gload_f32(params + 2);

  const __bf16* Q = Qb + (size_t)b * N_TOK * CK_DIM;
  const __bf16* K = Kb + (size_t)b * N_TOK * CK_DIM;
  const __bf16* V = Vbt + (size_t)b * C_DIM * N_TOK;

  const __bf16* qrow = Q + (size_t)(n0 + l16) * CK_DIM;
  v16bf a0 = load_a_g(qrow + 16 * half);
  v16bf a1 = load_a_g(qrow + 32 + 16 * half);

  float rmax[8];
#pragma unroll
  for (int r = 0; r < 8; ++r)
    rmax[r] = gload_f32(rowmax + (size_t)b * N_TOK + n0 + r + 8 * half);

  float rsum[8] = {0, 0, 0, 0, 0, 0, 0, 0};
  v8f acc[4] = {};

  for (int m0 = 0; m0 < N_TOK; m0 += MB) {
    // ---- TDM: stage V^T slab into LDS (wave 0 issues; EXEC is ignored) ----
    if (w == 0) {
      uint64_t gaddr  = (uint64_t)(uintptr_t)(V + m0);   // tile start: row c=0, col m0
      unsigned ldsoff = (unsigned)(uintptr_t)v_lds;
      u32x4 g0; i32x8 g1; i32x4 g2 = {0, 0, 0, 0}, g3 = {0, 0, 0, 0};
      g0[0] = 1u;                                                   // count=1 (valid load D#)
      g0[1] = ldsoff;                                               // lds_addr
      g0[2] = (unsigned)(gaddr & 0xffffffffu);                      // global_addr[31:0]
      g0[3] = (unsigned)((gaddr >> 32) & 0x1ffffffu) | (2u << 30);  // addr[56:32] | type=2
      g1[0] = (int)(1u << 16);                      // data_size = 2 bytes
      g1[1] = (int)((unsigned)N_TOK << 16);         // tensor_dim0[15:0]  (row length N)
      g1[2] = (int)((unsigned)C_DIM << 16);         // tensor_dim1[15:0]  (C rows)
      g1[3] = (int)((unsigned)MB << 16);            // tile_dim0 = MB (m window)
      g1[4] = C_DIM;                                // tile_dim1 = all C rows
      g1[5] = N_TOK;                                // tensor_dim0_stride[31:0]
      g1[6] = 0; g1[7] = 0;
#if defined(__clang_major__) && __clang_major__ >= 23
      i32x8 g4 = {0, 0, 0, 0, 0, 0, 0, 0};
      __builtin_amdgcn_tensor_load_to_lds(g0, g1, g2, g3, g4, 0);
#else
      __builtin_amdgcn_tensor_load_to_lds(g0, g1, g2, g3, 0);
#endif
    }

    // ---- overlapped: S tile for this wave's 16 m-columns -> quantized P ----
    {
      const int mw = m0 + w * 16;
      const __bf16* krow = K + (size_t)(mw + l16) * CK_DIM;
      v16bf b0 = gload_v16bf(krow + 16 * half);
      v16bf b1 = gload_v16bf(krow + 32 + 16 * half);
      v8f s = {};
      s = wmma_bf16(a0, b0, s);
      s = wmma_bf16(a1, b1, s);
#pragma unroll
      for (int r = 0; r < 8; ++r) {
        float e  = __builtin_exp2f((s[r] - rmax[r]) * 1.44269504f);
        float eq = __builtin_rintf(scale * e - zp);
        eq = fminf(fmaxf(eq, 0.0f), 255.0f);
        float ep = (eq + zp) * inv_scale;           // dequantized exponential
        rsum[r] += ep;
        p_lds[(size_t)(r + 8 * half) * MB + w * 16 + l16] = (__bf16)ep;
      }
    }

    if (w == 0) __builtin_amdgcn_s_wait_tensorcnt(0);
    __syncthreads();

    // ---- acc += P(16 x MB) * V(MB x 64-strip of this wave) ----
#pragma unroll
    for (int kc = 0; kc < MB / 32; ++kc) {
      v16bf pa = load_a_l(p_lds + (size_t)l16 * MB + kc * 32 + 16 * half);
#pragma unroll
      for (int ct = 0; ct < 4; ++ct) {
        const int c = w * 64 + ct * 16 + l16;
        // contiguous 16 m-values from the transposed V slab
        v16bf vb = *(const v16bf*)(v_lds + (size_t)c * MB + kc * 32 + 16 * half);
        acc[ct] = wmma_bf16(pa, vb, acc[ct]);
      }
    }
    __syncthreads();
  }

  // ---- combine per-wave row sums and normalize ----
#pragma unroll
  for (int m = 1; m < 16; m <<= 1)
#pragma unroll
    for (int r = 0; r < 8; ++r)
      rsum[r] += __shfl_xor(rsum[r], m, 32);
  if (l16 == 0)
#pragma unroll
    for (int r = 0; r < 8; ++r)
      rs_lds[w * 16 + r + 8 * half] = rsum[r];
  __syncthreads();
  if (tid < 16) {
    float t = 0.0f;
#pragma unroll
    for (int ww = 0; ww < 8; ++ww) t += rs_lds[ww * 16 + tid];
    inv_rs[tid] = 1.0f / t;
  }
  __syncthreads();

  __bf16* ob = attO + (size_t)b * N_TOK * C_DIM;      // [N][C]
#pragma unroll
  for (int ct = 0; ct < 4; ++ct) {
    const int c = w * 64 + ct * 16 + l16;
#pragma unroll
    for (int r = 0; r < 8; ++r) {
      int row = n0 + r + 8 * half;
      gstore_bf16(ob + (size_t)row * C_DIM + c, (__bf16)(acc[ct][r] * inv_rs[r + 8 * half]));
    }
  }
}

// ---------------------------------------------------------------------------
// K5: out = gamma * (wo @ attO^T + bo) + x   (fp32 output, [B][C][N] layout)
// ---------------------------------------------------------------------------
__global__ void outproj_kernel(const __bf16* __restrict__ attO, const float* __restrict__ wo,
                               const float* __restrict__ bo, const float* __restrict__ gamma,
                               const float* __restrict__ x, float* __restrict__ out) {
  const int lane = threadIdx.x & 31;
  const int half = lane >> 4, l16 = lane & 15;
  const int n0 = blockIdx.x * 16;
  const int o0 = blockIdx.y * 16;
  const int b  = blockIdx.z;

  const float*  worow = wo + (size_t)(o0 + l16) * C_DIM;               // A row = out chan
  const __bf16* arow  = attO + ((size_t)b * N_TOK + n0 + l16) * C_DIM; // B col = token

  v8f acc = {};
  for (int c0 = 0; c0 < C_DIM; c0 += 32) {
    v16bf a  = load_a_gf32(worow + c0 + 16 * half);
    v16bf bm = gload_v16bf(arow + c0 + 16 * half);                     // contiguous in c
    acc = wmma_bf16(a, bm, acc);
  }
  const float g = gload_f32(gamma);
  const int n = n0 + l16;
#pragma unroll
  for (int r = 0; r < 8; ++r) {
    int o = o0 + r + 8 * half;
    size_t idx = ((size_t)b * C_DIM + o) * N_TOK + n;
    gstore_f32(out + idx, g * (acc[r] + gload_f32(bo + o)) + gload_f32(x + idx));
  }
}

// ---------------------------------------------------------------------------
extern "C" void kernel_launch(void* const* d_in, const int* in_sizes, int n_in,
                              void* d_out, int out_size, void* d_ws, size_t ws_size,
                              hipStream_t stream) {
  (void)in_sizes; (void)n_in; (void)out_size; (void)ws_size;
  const float* x  = (const float*)d_in[0];
  const float* wq = (const float*)d_in[1];
  const float* bq = (const float*)d_in[2];
  const float* wk = (const float*)d_in[3];
  const float* bk = (const float*)d_in[4];
  const float* wv = (const float*)d_in[5];
  const float* bv = (const float*)d_in[6];
  const float* wo = (const float*)d_in[7];
  const float* bo = (const float*)d_in[8];
  const float* gm = (const float*)d_in[9];
  float* out = (float*)d_out;

  char* ws = (char*)d_ws;
  size_t off = 0;
  auto carve = [&](size_t bytes) -> void* {
    void* p = ws + off;
    off += (bytes + 255) & ~(size_t)255;
    return p;
  };
  __bf16* Qb   = (__bf16*)carve((size_t)B_DIM * N_TOK * CK_DIM * 2);
  __bf16* Kb   = (__bf16*)carve((size_t)B_DIM * N_TOK * CK_DIM * 2);
  __bf16* Vbt  = (__bf16*)carve((size_t)B_DIM * C_DIM * N_TOK * 2);
  __bf16* attO = (__bf16*)carve((size_t)B_DIM * N_TOK * C_DIM * 2);
  float*  rmax = (float*)carve((size_t)B_DIM * N_TOK * 4);
  float*  rmin = (float*)carve((size_t)B_DIM * N_TOK * 4);
  float*  prm  = (float*)carve(64);

  dim3 gp(N_TOK / 16, B_DIM);
  proj_kernel<<<gp, 256, 0, stream>>>(x, wq, bq, wk, bk, wv, bv, Qb, Kb, Vbt);

  dim3 gs(N_TOK / 16, B_DIM);
  rowstat_kernel<<<gs, 32, 0, stream>>>(Qb, Kb, rmax, rmin);

  quantparam_kernel<<<1, 256, 0, stream>>>(rmax, rmin, prm);

  size_t smem = (size_t)SMEM_V + SMEM_P + (8 * 16 + 16) * sizeof(float);
  (void)hipFuncSetAttribute(reinterpret_cast<const void*>(flash_kernel),
                            hipFuncAttributeMaxDynamicSharedMemorySize, (int)smem);
  dim3 gf(N_TOK / 16, B_DIM);
  flash_kernel<<<gf, 256, smem, stream>>>(Qb, Kb, Vbt, rmax, prm, attO);

  dim3 go(N_TOK / 16, C_DIM / 16, B_DIM);
  outproj_kernel<<<go, 32, 0, stream>>>(attO, wo, bo, gm, x, out);
}
